// FSM_44805098831977
// MI455X (gfx1250) — matrix operand order repaired and checked by reading, not compiled, not run on hardware
//
#include <hip/hip_runtime.h>

// ---------------- types / constants ----------------
typedef __attribute__((ext_vector_type(16))) __bf16          v16bf;
typedef __attribute__((ext_vector_type(16))) unsigned short  u16x16;
typedef __attribute__((ext_vector_type(8)))  float           v8f;

#define BB   8
#define CIN  40
#define COUT 40
#define HH   256
#define WW   256
#define KREAL 360                 // 9 taps * 40 channels
#define KPAD  384                 // 12 * 32
#define NKK   (KPAD / 32)         // 12 K-steps
#define NT    3                   // cout tiles (48 padded)
#define TILE_M 32                 // spatial pixels per workgroup (2 waves * 16)
#define NW    2                   // waves per block

#define WPACK_ELEMS (NKK * NT * 32 * 16)   // 18,432 bf16 per weight tensor
#define SW_BYTES    (WPACK_ELEMS * 2)      // 36,864 B
#define SA_ELEMS    (KPAD * TILE_M)        // 12,288 bf16
#define SA_BYTES    (SA_ELEMS * 2)         // 24,576 B
#define SMEM_BYTES  (SW_BYTES + SA_BYTES)  // 61,440 B

#define BIG_ELEMS   (BB * CIN * HH * WW)   // 20,971,520
#define SMALL_ELEMS (BB * CIN * 32 * 32)   // 327,680

__device__ __forceinline__ unsigned short f2bf(float f) {
  unsigned u = __float_as_uint(f);
  u += 0x7FFFu + ((u >> 16) & 1u);         // round-to-nearest-even
  return (unsigned short)(u >> 16);
}
__device__ __forceinline__ float gelu_f(float x) {
  return 0.5f * x * (1.0f + erff(x * 0.70710678118654752f));
}

// ---------------- weight pre-pack: OIHW f32 -> fragment-ordered bf16 ----------------
// target t: e = t&15, lane = (t>>4)&31, nt = (t>>9)%3, kk = (t>>9)/3
// B 32x16 layout: lanes 0-15 hold K {0-7,16-23}, lanes 16-31 hold K {8-15,24-31} (2 per VGPR)
__global__ __launch_bounds__(256) void pack_w_k(
    const float* __restrict__ wa, const float* __restrict__ wb,
    const float* __restrict__ wc, const float* __restrict__ wd,
    unsigned short* __restrict__ out) {
  int idx = blockIdx.x * 256 + threadIdx.x;
  if (idx >= 4 * WPACK_ELEMS) return;
  int which = idx / WPACK_ELEMS;
  int t = idx % WPACK_ELEMS;
  const float* w = (which == 0) ? wa : (which == 1) ? wb : (which == 2) ? wc : wd;
  int e = t & 15, lane = (t >> 4) & 31, r = t >> 9;
  int nt = r % NT, kk = r / NT;
  int hi = lane >> 4, lo = lane & 15;
  int k = kk * 32 + ((e & 8) << 1) + hi * 8 + (e & 7);
  int n = nt * 16 + lo;
  float v = 0.f;
  if (k < KREAL && n < COUT) {
    int ci = k % CIN, tap = k / CIN;            // k = tap*CIN + ci
    v = w[(n * CIN + ci) * 9 + tap];
  }
  out[idx] = f2bf(v);
}

// ---------------- WMMA implicit-GEMM 3x3 conv ----------------
// M = 32 spatial (2 waves x 16), N = 48 cout (3 tiles), K = 384
// EPI: 0 = gelu, 1 = gelu + residual, 2 = relu, 3 = identity
template <int EPI>
__global__ __launch_bounds__(64) void conv3x3_wmma(
    const float* __restrict__ in, const unsigned short* __restrict__ wpack,
    const float* __restrict__ bias, const float* res,
    float* __restrict__ out) {
  extern __shared__ __align__(16) char smem[];
  unsigned short* sWf = (unsigned short*)smem;              // 36,864 B
  unsigned short* sAf = (unsigned short*)(smem + SW_BYTES); // 24,576 B

  const int tid = threadIdx.x;
  const int w0  = blockIdx.x * TILE_M;
  const int h   = blockIdx.y;
  const int b   = blockIdx.z;

  // ---- async DMA: packed weight tile straight into LDS (ASYNCcnt path) ----
  {
    unsigned ldsBase = (unsigned)(size_t)(void*)sWf;
    unsigned long long g = (unsigned long long)(size_t)wpack;
    for (int c = tid; c < SW_BYTES / 16; c += 64) {
      unsigned lds = ldsBase + (unsigned)c * 16u;
      unsigned long long ga = g + (unsigned long long)c * 16u;
      asm volatile("global_load_async_to_lds_b128 %0, %1, off"
                   :: "v"(lds), "v"(ga) : "memory");
    }
  }

  // ---- im2col directly into A-fragment order (coalesced: m fastest) ----
  // A 16x32 layout: lane<16 -> K {0-7,16-23}; lane>=16 -> K {8-15,24-31}
  const float* inb = in + (size_t)b * CIN * HH * WW;
  for (int s = tid; s < KPAD * TILE_M; s += 64) {
    int m  = s & (TILE_M - 1);
    int ke = s >> 5;
    float v = 0.f;
    if (ke < KREAL) {
      int ci = ke % CIN, tap = ke / CIN;
      int kh = tap / 3, kw = tap % 3;
      int ih = h + kh - 1, iw = w0 + m + kw - 1;
      if (ih >= 0 && ih < HH && iw >= 0 && iw < WW)
        v = inb[(ci * HH + ih) * WW + iw];
    }
    int kk = ke >> 5, krem = ke & 31;
    int hi = (krem >> 3) & 1;
    int e  = (((krem & 16) >> 2) + ((krem & 7) >> 1)) * 2 + (krem & 1);
    int lane = hi * 16 + (m & 15);
    int wave = m >> 4;
    sAf[((kk * NW + wave) * 32 + lane) * 16 + e] = f2bf(v);
  }
  __builtin_prefetch(inb + h * WW + w0, 0, 1);   // global_prefetch_b8
  asm volatile("s_wait_asynccnt 0x0" ::: "memory");
  __syncthreads();

  // ---- K-loop: contiguous 32B LDS fragment loads + 3 WMMA per step ----
  const int lane  = tid & 31;
  const int wave  = tid >> 5;
  const int hi    = lane >> 4;
  const int lo    = lane & 15;
  const int mbase = wave * 16;

  const u16x16* pW = (const u16x16*)sWf;
  const u16x16* pA = (const u16x16*)sAf;
  union LdU { u16x16 s; v16bf v; };

  v8f acc0 = {}, acc1 = {}, acc2 = {};
  for (int kk = 0; kk < NKK; ++kk) {
    LdU a, b0, b1, b2;
    a.s  = pA[(kk * NW + wave) * 32 + lane];
    b0.s = pW[(kk * NT + 0) * 32 + lane];
    b1.s = pW[(kk * NT + 1) * 32 + lane];
    b2.s = pW[(kk * NT + 2) * 32 + lane];
    acc0 = __builtin_amdgcn_wmma_f32_16x16x32_bf16(false, a.v, false, b0.v, (short)0, acc0, false, false);
    acc1 = __builtin_amdgcn_wmma_f32_16x16x32_bf16(false, a.v, false, b1.v, (short)0, acc1, false, false);
    acc2 = __builtin_amdgcn_wmma_f32_16x16x32_bf16(false, a.v, false, b2.v, (short)0, acc2, false, false);
  }

  // ---- epilogue: C/D layout -> lane lo = cout, VGPR r -> m = r + 8*hi ----
  float* outb = out + (size_t)b * COUT * HH * WW;
  const float* resb = (EPI == 1) ? (res + (size_t)b * COUT * HH * WW) : nullptr;
#pragma unroll
  for (int nt = 0; nt < NT; ++nt) {
    v8f acc = (nt == 0) ? acc0 : ((nt == 1) ? acc1 : acc2);
    int co = nt * 16 + lo;
    if (co < COUT) {
      float bv = bias[co];
#pragma unroll
      for (int r = 0; r < 8; ++r) {
        int m = mbase + r + 8 * hi;
        size_t oidx = (size_t)(co * HH + h) * WW + (w0 + m);
        float v = acc[r] + bv;
        if (EPI == 0)      v = gelu_f(v);
        else if (EPI == 1) v = gelu_f(v) + resb[oidx];
        else if (EPI == 2) v = fmaxf(v, 0.f);
        outb[oidx] = v;
      }
    }
  }
}

// ---------------- depthwise dilated stride-8 conv -> 32x32 ----------------
__global__ __launch_bounds__(256) void dwconv_k(
    const float* __restrict__ in, const float* __restrict__ w,
    const float* __restrict__ bias, float* __restrict__ out) {
  int idx = blockIdx.x * blockDim.x + threadIdx.x;
  if (idx >= SMALL_ELEMS) return;
  int ow = idx & 31, oh = (idx >> 5) & 31;
  int c  = (idx >> 10) % CIN, b = idx / (1024 * CIN);
  const float* ip = in + (size_t)(b * CIN + c) * HH * WW;
  float s = bias[c];
#pragma unroll
  for (int kh = 0; kh < 3; ++kh) {
    int ih = oh * 8 - 2 + 2 * kh;
    if (ih < 0 || ih >= HH) continue;
#pragma unroll
    for (int kw = 0; kw < 3; ++kw) {
      int iw = ow * 8 - 2 + 2 * kw;
      if (iw < 0 || iw >= WW) continue;
      s += w[c * 9 + kh * 3 + kw] * ip[ih * WW + iw];
    }
  }
  out[idx] = s;
}

// ---------------- per-8x8-block IDCT (Re(ifft) rows then cols) ----------------
__device__ __constant__ float COS8[8] = {
  1.f, 0.70710678118654752f, 0.f, -0.70710678118654752f,
  -1.f, -0.70710678118654752f, 0.f, 0.70710678118654752f };

__global__ __launch_bounds__(256) void idct_k(
    const float* __restrict__ in, float* __restrict__ out) {
  int idx = blockIdx.x * blockDim.x + threadIdx.x;
  if (idx >= SMALL_ELEMS) return;
  int l = idx & 31, k = (idx >> 5) & 31;
  int c = (idx >> 10) % CIN, b = idx / (1024 * CIN);
  int bh = k & ~7, bw = l & ~7, kr = k & 7, lr = l & 7;
  const float* ip = in + (size_t)(b * CIN + c) * 1024;
  float s = 0.f;
#pragma unroll
  for (int m = 0; m < 8; ++m) {
    float rk = COS8[(kr * m) & 7];
#pragma unroll
    for (int n = 0; n < 8; ++n)
      s += rk * COS8[(lr * n) & 7] * ip[(bh + m) * 32 + (bw + n)];
  }
  out[idx] = s * 0.125f;   // (1/sqrt(8))^2
}

// ---------------- 1x1 conv + sigmoid (gate branch) ----------------
__global__ __launch_bounds__(256) void conv1x1_sig_k(
    const float* __restrict__ t, const float* __restrict__ w,
    const float* __restrict__ bias, float* __restrict__ out) {
  int idx = blockIdx.x * blockDim.x + threadIdx.x;
  if (idx >= BIG_ELEMS) return;
  int hw = idx % (HH * WW);
  int co = (idx / (HH * WW)) % COUT;
  int b  = idx / (HH * WW * COUT);
  const float* tb = t + (size_t)b * CIN * HH * WW + hw;
  float s = bias[co];
#pragma unroll 4
  for (int ci = 0; ci < CIN; ++ci)
    s += w[co * CIN + ci] * tb[(size_t)ci * HH * WW];
  out[idx] = 1.f / (1.f + expf(-s));
}

// ---------------- bilinear 32->256 upsample (half-pixel) * gate ----------------
__global__ __launch_bounds__(256) void upmul_k(
    const float* __restrict__ sm, const float* __restrict__ wgt,
    float* __restrict__ out) {
  int idx = blockIdx.x * blockDim.x + threadIdx.x;
  if (idx >= BIG_ELEMS) return;
  int x = idx & 255, y = (idx >> 8) & 255;
  int c = (idx >> 16) % CIN, b = idx / (65536 * CIN);
  const float* sp = sm + (size_t)(b * CIN + c) * 1024;
  float sy = (y + 0.5f) * 0.125f - 0.5f;
  float sx = (x + 0.5f) * 0.125f - 0.5f;
  int y0 = (int)floorf(sy), x0 = (int)floorf(sx);
  float fy = sy - (float)y0, fx = sx - (float)x0;
  int y0c = min(31, max(0, y0)),     x0c = min(31, max(0, x0));
  int y1c = min(31, max(0, y0 + 1)), x1c = min(31, max(0, x0 + 1));
  float v00 = sp[y0c * 32 + x0c], v01 = sp[y0c * 32 + x1c];
  float v10 = sp[y1c * 32 + x0c], v11 = sp[y1c * 32 + x1c];
  float v = (1.f - fy) * ((1.f - fx) * v00 + fx * v01)
          +        fy  * ((1.f - fx) * v10 + fx * v11);
  out[idx] = v * wgt[idx];
}

// ---------------- host launch ----------------
extern "C" void kernel_launch(void* const* d_in, const int* in_sizes, int n_in,
                              void* d_out, int out_size, void* d_ws, size_t ws_size,
                              hipStream_t stream) {
  const float* x      = (const float*)d_in[0];
  const float* conv_w = (const float*)d_in[1];
  const float* conv_b = (const float*)d_in[2];
  const float* ddct_w = (const float*)d_in[3];
  const float* ddct_b = (const float*)d_in[4];
  const float* dctc_w = (const float*)d_in[5];
  const float* dctc_b = (const float*)d_in[6];
  const float* w1_w   = (const float*)d_in[7];
  const float* w1_b   = (const float*)d_in[8];
  const float* w3_w   = (const float*)d_in[9];
  const float* w3_b   = (const float*)d_in[10];
  const float* after_w = (const float*)d_in[11];
  const float* after_b = (const float*)d_in[12];

  const size_t BIG = (size_t)BIG_ELEMS * sizeof(float);    // 83.9 MB
  const size_t SML = (size_t)SMALL_ELEMS * sizeof(float);  // 1.31 MB
  float* bufA = (float*)d_ws;                              // dct_feat, later wgt
  float* bufB = (float*)((char*)d_ws + BIG);               // dd
  float* bufC = (float*)((char*)d_ws + 2 * BIG);           // relu branch, later gated product
  float* s1   = (float*)((char*)d_ws + 3 * BIG);           // dct_feat_0 (32x32)
  float* s2   = (float*)((char*)d_ws + 3 * BIG + SML);     // out_0 idct (32x32)
  unsigned short* wpack = (unsigned short*)((char*)d_ws + 3 * BIG + 2 * SML);

  dim3 cgrid(WW / TILE_M, HH, BB);   // (8, 256, 8)
  dim3 cblk(64);
  int bigBlocks = (BIG_ELEMS + 255) / 256;
  int smlBlocks = (SMALL_ELEMS + 255) / 256;

  // 0) pack all four conv weights into WMMA fragment order (bf16)
  pack_w_k<<<(4 * WPACK_ELEMS + 255) / 256, 256, 0, stream>>>(
      conv_w, ddct_w, w1_w, after_w, wpack);

  unsigned short* wp_conv  = wpack;
  unsigned short* wp_ddct  = wpack + WPACK_ELEMS;
  unsigned short* wp_w1    = wpack + 2 * WPACK_ELEMS;
  unsigned short* wp_after = wpack + 3 * WPACK_ELEMS;

  // 1) dct_feat = gelu(conv3x3(x))
  conv3x3_wmma<0><<<cgrid, cblk, SMEM_BYTES, stream>>>(x, wp_conv, conv_b, nullptr, bufA);
  // 2) dd = gelu(conv3x3(dct_feat)) + dct_feat
  conv3x3_wmma<1><<<cgrid, cblk, SMEM_BYTES, stream>>>(bufA, wp_ddct, ddct_b, bufA, bufB);
  // 3) depthwise dilated stride-8 conv -> 32x32
  dwconv_k<<<smlBlocks, 256, 0, stream>>>(bufB, dctc_w, dctc_b, s1);
  // 4) per-block IDCT
  idct_k<<<smlBlocks, 256, 0, stream>>>(s1, s2);
  // 5) gate branch: relu(conv3x3(x))
  conv3x3_wmma<2><<<cgrid, cblk, SMEM_BYTES, stream>>>(x, wp_w1, w1_b, nullptr, bufC);
  // 6) wgt = sigmoid(conv1x1(.))
  conv1x1_sig_k<<<bigBlocks, 256, 0, stream>>>(bufC, w3_w, w3_b, bufA);
  // 7) gated = upsample(out_0) * wgt
  upmul_k<<<bigBlocks, 256, 0, stream>>>(s2, bufA, bufC);
  // 8) out = conv3x3(gated)
  conv3x3_wmma<3><<<cgrid, cblk, SMEM_BYTES, stream>>>(bufC, after_w ? wp_after : wp_after, after_b, nullptr, (float*)d_out);
}
